// DualEncoderSCFM_29033978921577
// MI455X (gfx1250) — matrix operand order repaired and checked by reading, not compiled, not run on hardware
//
#include <hip/hip_runtime.h>
#include <math.h>

typedef __attribute__((ext_vector_type(16))) _Float16 v16h;
typedef __attribute__((ext_vector_type(8)))  float    v8f;

#define TM 64
#define TN 64
#define TK 32
#define KPAD 8   // LDS row stride = TK+KPAD = 40 halves = 80B (16B aligned)

enum { GF_TA=1, GF_TB=2, GF_BIAS=4, GF_RELU=8, GF_GELU=16, GF_RESID=32, GF_RS=64 };

struct GemmP {
  const float* A; const float* B; const float* bias; const float* resid; const float* rs;
  float* C;
  int M, N, K;
  int lda, ldb, ldc;
  long sA, sB, sC, sResid, sRS;
  float alpha;
  int flags;
};

// ---- WMMA GEMM: C = act( alpha*A (*) B + bias ) [*rowscale] [+resid] ----
// Block tile 64x64, 8 waves; each wave: one 16-row slab x 32 cols (2 accumulators).
template <bool TA, bool TB>
__global__ __launch_bounds__(256) void k_gemm(GemmP p) {
  __shared__ __align__(16) _Float16 As[TM][TK + KPAD];   // [m][k]
  __shared__ __align__(16) _Float16 Bs[TN][TK + KPAD];   // [n][k] (col-major tile)
  int z = blockIdx.z;
  const float* A = p.A + (long)z * p.sA;
  const float* B = p.B + (long)z * p.sB;
  float* C = p.C + (long)z * p.sC;
  int bm0 = blockIdx.y * TM, bn0 = blockIdx.x * TN;
  int t = threadIdx.x;
  int w = t >> 5, lane = t & 31;
  int wr = w & 3, wc = w >> 2;
  bool hi = lane >= 16;
  int l15 = lane & 15;
  v8f acc0 = {}, acc1 = {};
  int ktiles = (p.K + TK - 1) / TK;
  bool fullMN = (bm0 + TM <= p.M) && (bn0 + TN <= p.N);
  for (int kt = 0; kt < ktiles; ++kt) {
    int k0 = kt * TK;
    bool full = fullMN && (k0 + TK <= p.K);
    if (full) {
      if (!TA) {
        int row = t >> 2, c0 = (t & 3) * 8;
        const float* src = A + (long)(bm0 + row) * p.lda + k0 + c0;
#pragma unroll
        for (int j = 0; j < 8; ++j) As[row][c0 + j] = (_Float16)(src[j] * p.alpha);
        __builtin_prefetch(src + TK, 0, 0);
      } else {
        int k = t >> 3, r0 = (t & 7) * 8;
        const float* src = A + (long)(k0 + k) * p.lda + bm0 + r0;
#pragma unroll
        for (int j = 0; j < 8; ++j) As[r0 + j][k] = (_Float16)(src[j] * p.alpha);
        __builtin_prefetch(src + (long)TK * p.lda, 0, 0);
      }
      if (!TB) {
        int k = t >> 3, c0 = (t & 7) * 8;
        const float* src = B + (long)(k0 + k) * p.ldb + bn0 + c0;
#pragma unroll
        for (int j = 0; j < 8; ++j) Bs[c0 + j][k] = (_Float16)src[j];
        __builtin_prefetch(src + (long)TK * p.ldb, 0, 0);
      } else {
        int c = t >> 2, kk0 = (t & 3) * 8;
        const float* src = B + (long)(bn0 + c) * p.ldb + k0 + kk0;
#pragma unroll
        for (int j = 0; j < 8; ++j) Bs[c][kk0 + j] = (_Float16)src[j];
        __builtin_prefetch(src + TK, 0, 0);
      }
    } else {
      if (!TA) {
        int row = t >> 2, c0 = (t & 3) * 8;
        int gm = bm0 + row;
#pragma unroll
        for (int j = 0; j < 8; ++j) {
          int gk = k0 + c0 + j;
          float v = (gm < p.M && gk < p.K) ? A[(long)gm * p.lda + gk] : 0.f;
          As[row][c0 + j] = (_Float16)(v * p.alpha);
        }
      } else {
        int k = t >> 3, r0 = (t & 7) * 8;
        int gk = k0 + k;
#pragma unroll
        for (int j = 0; j < 8; ++j) {
          int gm = bm0 + r0 + j;
          float v = (gm < p.M && gk < p.K) ? A[(long)gk * p.lda + gm] : 0.f;
          As[r0 + j][k] = (_Float16)(v * p.alpha);
        }
      }
      if (!TB) {
        int k = t >> 3, c0 = (t & 7) * 8;
        int gk = k0 + k;
#pragma unroll
        for (int j = 0; j < 8; ++j) {
          int gn = bn0 + c0 + j;
          float v = (gk < p.K && gn < p.N) ? B[(long)gk * p.ldb + gn] : 0.f;
          Bs[c0 + j][k] = (_Float16)v;
        }
      } else {
        int c = t >> 2, kk0 = (t & 3) * 8;
        int gn = bn0 + c;
#pragma unroll
        for (int j = 0; j < 8; ++j) {
          int gk = k0 + kk0 + j;
          float v = (gn < p.N && gk < p.K) ? B[(long)gn * p.ldb + gk] : 0.f;
          Bs[c][kk0 + j] = (_Float16)v;
        }
      }
    }
    __syncthreads();
    // Vectorized fragment loads (2x b128 per fragment).
    // A frag: elems 0..15 = K ka0..ka0+15 of row ar.
    // B frag (col-major LDS): elems 0..15 = K kb0..kb0+15 of column bc.
    union V16 { uint4 q[2]; v16h v; };
    V16 av, bv0, bv1;
    int ar = wr * 16 + l15;
    int ka0 = hi ? 16 : 0;
    av.q[0] = *(const uint4*)&As[ar][ka0];
    av.q[1] = *(const uint4*)&As[ar][ka0 + 8];
    int bc = wc * 32 + l15;
    int kb0 = hi ? 16 : 0;
    bv0.q[0] = *(const uint4*)&Bs[bc][kb0];
    bv0.q[1] = *(const uint4*)&Bs[bc][kb0 + 8];
    bv1.q[0] = *(const uint4*)&Bs[bc + 16][kb0];
    bv1.q[1] = *(const uint4*)&Bs[bc + 16][kb0 + 8];
    acc0 = __builtin_amdgcn_wmma_f32_16x16x32_f16(false, av.v, false, bv0.v, (short)0, acc0, false, false);
    acc1 = __builtin_amdgcn_wmma_f32_16x16x32_f16(false, av.v, false, bv1.v, (short)0, acc1, false, false);
    __syncthreads();
  }
  // epilogue: C/D layout lane(0-15)=N, VGPR i -> M = i (+8 for hi lanes)
  int row0 = bm0 + wr * 16 + (hi ? 8 : 0);
  int colA = bn0 + wc * 32 + l15;
  v8f* accs[2] = {&acc0, &acc1};
#pragma unroll
  for (int half = 0; half < 2; ++half) {
    int col = colA + half * 16;
    if (col < p.N) {
      v8f& acc = *accs[half];
#pragma unroll
      for (int i = 0; i < 8; ++i) {
        int row = row0 + i;
        if (row < p.M) {
          float v = acc[i];
          if (p.flags & GF_BIAS) v += p.bias[col];
          if (p.flags & GF_RS)   v *= p.rs[(long)z * p.sRS + row];
          if (p.flags & GF_GELU) v = 0.5f * v * (1.f + erff(v * 0.70710678f));
          if (p.flags & GF_RELU) v = fmaxf(v, 0.f);
          if (p.flags & GF_RESID) v += p.resid[(long)z * p.sResid + (long)row * p.ldc + col];
          C[(long)row * p.ldc + col] = v;
        }
      }
    }
  }
}

// ---------------- small kernels ----------------
__global__ void k_zero(float* p, long n) {
  long i = (long)blockIdx.x * blockDim.x + threadIdx.x;
  if (i < n) p[i] = 0.f;
}
__global__ void k_add(float* a, const float* b, long n) {
  long i = (long)blockIdx.x * blockDim.x + threadIdx.x;
  if (i < n) a[i] += b[i];
}
__global__ void k_mask_renorm(const float* mask, float* out) {
  __shared__ float red[256];
  int t = threadIdx.x;
  float v = 0.f;
  if (t < 200) { float m = mask[t]; v = m * m; }
  red[t] = v; __syncthreads();
  for (int s = 128; s > 0; s >>= 1) { if (t < s) red[t] += red[t + s]; __syncthreads(); }
  float sc = fminf(1.f, 1.f / (sqrtf(red[0]) + 1e-7f));
  if (t < 200) out[t] = mask[t] * sc;
}
__global__ void k_token_embed(const float* x, const float* w1, const float* b1,
                              const float* w2, const float* b2, const float* mrow,
                              const float* g, const float* bta, float* out, int n) {
  int gi = blockIdx.x; if (gi >= n) return;
  __shared__ float hid[50];
  __shared__ float red[256];
  int t = threadIdx.x;
  float xv = x[gi];
  if (t < 50) hid[t] = fmaxf(0.f, xv * w1[t] + b1[t]);
  __syncthreads();
  float val = 0.f;
  if (t < 200) {
    float s = b2[t];
    for (int hh = 0; hh < 50; ++hh) s += hid[hh] * w2[hh * 200 + t];
    float m = (xv <= -1.0f) ? 1.f : 0.f;
    val = (1.f - m) * s + m * mrow[t];
  }
  red[t] = (t < 200) ? val : 0.f; __syncthreads();
  for (int s = 128; s > 0; s >>= 1) { if (t < s) red[t] += red[t + s]; __syncthreads(); }
  float mu = red[0] / 200.f; __syncthreads();
  float dv = (t < 200) ? (val - mu) : 0.f;
  red[t] = dv * dv; __syncthreads();
  for (int s = 128; s > 0; s >>= 1) { if (t < s) red[t] += red[t + s]; __syncthreads(); }
  float inv = rsqrtf(red[0] / 200.f + 1e-5f);
  if (t < 200) out[(long)gi * 200 + t] = (val - mu) * inv * g[t] + bta[t];
}
__global__ void k_layernorm(const float* src, float* dst, const float* g, const float* b,
                            int n, int d) {
  int r = blockIdx.x; if (r >= n) return;
  const float* row = src + (long)r * d;
  __shared__ float red[256];
  int t = threadIdx.x;
  float s = 0.f;
  for (int j = t; j < d; j += 256) s += row[j];
  red[t] = s; __syncthreads();
  for (int k = 128; k > 0; k >>= 1) { if (t < k) red[t] += red[t + k]; __syncthreads(); }
  float mu = red[0] / d; __syncthreads();
  float vs = 0.f;
  for (int j = t; j < d; j += 256) { float dd = row[j] - mu; vs += dd * dd; }
  red[t] = vs; __syncthreads();
  for (int k = 128; k > 0; k >>= 1) { if (t < k) red[t] += red[t + k]; __syncthreads(); }
  float inv = rsqrtf(red[0] / d + 1e-5f);
  float* orow = dst + (long)r * d;
  for (int j = t; j < d; j += 256) orow[j] = (row[j] - mu) * inv * g[j] + b[j];
}
__global__ void k_deg(const int* ei, const float* w, float* deg, int e) {
  int i = blockIdx.x * blockDim.x + threadIdx.x;
  if (i < e) atomicAdd(&deg[ei[e + i]], w[i]);
}
__global__ void k_dis(const float* deg, float* dis, int n) {
  int i = blockIdx.x * blockDim.x + threadIdx.x;
  if (i < n) { float d = deg[i]; dis[i] = (d > 0.f) ? rsqrtf(d) : 0.f; }
}
__global__ void k_sgconv(const int* ei, const float* w, const float* dis, const float* table,
                         float* agg, int e) {
  int ed = blockIdx.x; if (ed >= e) return;
  int r = ei[ed], c = ei[e + ed];
  float norm = dis[r] * w[ed] * dis[c];
  int t = threadIdx.x;
  if (t < 200) atomicAdd(&agg[(long)c * 200 + t], norm * table[(long)r * 200 + t]);
}
__global__ void k_finish_embed(const float* pos, const float* go, float* posgo, float* xemb, long n) {
  long i = (long)blockIdx.x * blockDim.x + threadIdx.x;
  if (i < n) { float p = pos[i] + go[i]; posgo[i] = p; xemb[i] += p; }
}
__global__ void k_keys(const float* x, unsigned* keys, int n) {
  int i = blockIdx.x * blockDim.x + threadIdx.x;
  if (i < n) {
    unsigned u = __float_as_uint(x[i]);
    keys[i] = (u >> 31) ? ~u : (u | 0x80000000u);
  }
}
__global__ void k_radix_select(const unsigned* keys, int n, int K, unsigned* outTE) {
  __shared__ unsigned hist[256];
  __shared__ unsigned prefix_s;
  __shared__ int kwant_s;
  __shared__ unsigned gcount;
  int t = threadIdx.x;
  if (t == 0) { prefix_s = 0u; kwant_s = K; gcount = 0u; }
  __syncthreads();
  for (int pass = 0; pass < 4; ++pass) {
    int shift = 24 - 8 * pass;
    unsigned mask = (pass == 0) ? 0u : (0xFFFFFFFFu << (32 - 8 * pass));
    hist[t] = 0u; __syncthreads();
    unsigned prefix = prefix_s;
    for (int i = t; i < n; i += 256) {
      unsigned ky = keys[i];
      if ((ky & mask) == prefix) atomicAdd(&hist[(ky >> shift) & 0xFF], 1u);
    }
    __syncthreads();
    if (t == 0) {
      int kw = kwant_s; unsigned sel = 0u;
      for (int d = 255; d >= 0; --d) {
        unsigned c = hist[d];
        if ((int)c >= kw) { sel = (unsigned)d; break; }
        kw -= (int)c;
      }
      kwant_s = kw;
      prefix_s = prefix | (sel << shift);
    }
    __syncthreads();
  }
  unsigned T = prefix_s;
  unsigned local = 0u;
  for (int i = t; i < n; i += 256) if (keys[i] > T) local++;
  atomicAdd(&gcount, local);
  __syncthreads();
  if (t == 0) { outTE[0] = T; outTE[1] = (unsigned)(K - (int)gcount); }
}
__global__ void k_partition(const unsigned* keys, const unsigned* TE, int n,
                            int* top_idx, int* left_idx) {
  __shared__ int scan[256];
  __shared__ int topBase, leftBase, eqBase;
  int t = threadIdx.x;
  if (t == 0) { topBase = 0; leftBase = 0; eqBase = 0; }
  __syncthreads();
  unsigned T = TE[0]; int needEq = (int)TE[1];
  for (int base = 0; base < n; base += 256) {
    int i = base + t;
    int valid = i < n;
    unsigned ky = valid ? keys[i] : 0u;
    int g = valid && (ky > T);
    int e = valid && (ky == T);
    scan[t] = e; __syncthreads();
    for (int off = 1; off < 256; off <<= 1) {
      int v = scan[t]; if (t >= off) v += scan[t - off];
      __syncthreads(); scan[t] = v; __syncthreads();
    }
    int eIncl = scan[t], eTot = scan[255];
    int isTop = g || (e && (eqBase + eIncl - 1) < needEq);
    __syncthreads();
    scan[t] = isTop; __syncthreads();
    for (int off = 1; off < 256; off <<= 1) {
      int v = scan[t]; if (t >= off) v += scan[t - off];
      __syncthreads(); scan[t] = v; __syncthreads();
    }
    int tIncl = scan[t], tTot = scan[255];
    int validCnt = (n - base < 256) ? (n - base) : 256;
    if (valid) {
      if (isTop) top_idx[topBase + tIncl - 1] = i;
      else       left_idx[leftBase + (t + 1 - tIncl) - 1] = i;
    }
    __syncthreads();
    if (t == 0) { topBase += tTot; leftBase += validCnt - tTot; eqBase += eTot; }
    __syncthreads();
  }
}
__global__ void k_gather_rows(const float* src, const int* idx, float* dst, int rows, int d) {
  int r = blockIdx.x; if (r >= rows) return;
  int s = idx[r];
  for (int j = threadIdx.x; j < d; j += blockDim.x)
    dst[(long)r * d + j] = src[(long)s * d + j];
}
__global__ void k_scatter_rows(const float* src, const int* idx, float* dst, int rows, int d) {
  int r = blockIdx.x; if (r >= rows) return;
  int dd = idx[r];
  for (int j = threadIdx.x; j < d; j += blockDim.x)
    dst[(long)dd * d + j] = src[(long)r * d + j];
}
__global__ void k_row_sqsum(const float* A, int lda, long sA, float* out, int n, int d, float scale) {
  int z = blockIdx.y, r = blockIdx.x; if (r >= n) return;
  const float* row = A + (long)z * sA + (long)r * lda;
  __shared__ float red[256];
  int t = threadIdx.x;
  float s = 0.f;
  for (int j = t; j < d; j += 256) { float v = row[j]; s += v * v; }
  red[t] = s; __syncthreads();
  for (int k = 128; k > 0; k >>= 1) { if (t < k) red[t] += red[t + k]; __syncthreads(); }
  if (t == 0) out[(long)z * n + r] = scale * red[0];
}
__global__ void k_row_max(const float* A, int lda, long sA, float* out, int n, int d) {
  int z = blockIdx.y, r = blockIdx.x; if (r >= n) return;
  const float* row = A + (long)z * sA + (long)r * lda;
  __shared__ float red[256];
  int t = threadIdx.x;
  float m = -3.0e38f;
  for (int j = t; j < d; j += 256) m = fmaxf(m, row[j]);
  red[t] = m; __syncthreads();
  for (int k = 128; k > 0; k >>= 1) { if (t < k) red[t] = fmaxf(red[t], red[t + k]); __syncthreads(); }
  if (t == 0) out[(long)z * n + r] = red[0];
}
__global__ void k_head_max(const float* rowmax, float* out, int n) {
  int z = blockIdx.x;
  __shared__ float red[256];
  int t = threadIdx.x;
  float m = -3.0e38f;
  for (int i = t; i < n; i += 256) m = fmaxf(m, rowmax[(long)z * n + i]);
  red[t] = m; __syncthreads();
  for (int k = 128; k > 0; k >>= 1) { if (t < k) red[t] = fmaxf(red[t], red[t + k]); __syncthreads(); }
  if (t == 0) out[z] = red[0];
}
__global__ void k_feature_map(float* dd, const float* diag, const float* stab,
                              int n, int m, float ratio, int perRow, float keps) {
  int z = blockIdx.y;
  long i = (long)blockIdx.x * 256 + threadIdx.x;
  if (i >= (long)n * m) return;
  int r = (int)(i / m);
  float st = perRow ? stab[(long)z * n + r] : stab[z];
  long o = (long)z * n * m + i;
  dd[o] = ratio * (expf(dd[o] - diag[(long)z * n + r] - st) + keps);
}
__global__ void k_col_sum(const float* kp, float* ksum, int n, int m) {
  int z = blockIdx.y;
  int c = blockIdx.x * 256 + threadIdx.x;
  if (c >= m) return;
  const float* base = kp + (long)z * n * m;
  float s = 0.f;
  for (int r = 0; r < n; ++r) s += base[(long)r * m + c];
  ksum[(long)z * m + c] = s;
}
__global__ void k_dinv(const float* qp, const float* ksum, float* dinv, int n, int m) {
  int z = blockIdx.y, r = blockIdx.x; if (r >= n) return;
  const float* qr = qp + (long)z * n * m + (long)r * m;
  const float* ks = ksum + (long)z * m;
  __shared__ float red[256];
  int t = threadIdx.x;
  float s = 0.f;
  for (int j = t; j < m; j += 256) s += qr[j] * ks[j];
  red[t] = s; __syncthreads();
  for (int k = 128; k > 0; k >>= 1) { if (t < k) red[t] += red[t + k]; __syncthreads(); }
  if (t == 0) dinv[(long)z * n + r] = 1.f / red[0];
}

// ---------------- host side ----------------
static void gemm(hipStream_t st, const float* A, int lda, long sA,
                 const float* B, int ldb, long sB, float* C, int ldc, long sC,
                 int M, int N, int K, int batch, int flags,
                 const float* bias = nullptr, const float* resid = nullptr, long sResid = 0,
                 const float* rs = nullptr, long sRS = 0, float alpha = 1.f) {
  GemmP p{A, B, bias, resid, rs, C, M, N, K, lda, ldb, ldc, sA, sB, sC, sResid, sRS, alpha, flags};
  dim3 g((N + TN - 1) / TN, (M + TM - 1) / TM, batch);
  if (flags & GF_TA)      k_gemm<true,  false><<<g, 256, 0, st>>>(p);
  else if (flags & GF_TB) k_gemm<false, true ><<<g, 256, 0, st>>>(p);
  else                    k_gemm<false, false><<<g, 256, 0, st>>>(p);
}

struct LayerW {
  const float *ln1g, *ln1b, *wq, *wk, *wv, *outw, *outb, *ln2g, *ln2b, *ff1w, *ff1b, *ff2w, *ff2b;
};
struct Scratch {
  float *tln, *q, *k, *v, *qp, *kp, *diag, *stab, *stabh, *ksum, *dinv, *ctx, *om, *ff;
};

static void performer_layer(hipStream_t st, float* x, int n, int d, int h, int dh, int mfeat,
                            const float* proj, const LayerW& w, const Scratch& S) {
  int inner = h * dh;
  float dn = 1.f / sqrtf(sqrtf((float)dh));
  float ratio = 1.f / sqrtf((float)mfeat);
  long nm = (long)n * mfeat;
  k_layernorm<<<n, 256, 0, st>>>(x, S.tln, w.ln1g, w.ln1b, n, d);
  gemm(st, S.tln, d, 0, w.wq, inner, 0, S.q, inner, 0, n, inner, d, 1, 0);
  gemm(st, S.tln, d, 0, w.wk, inner, 0, S.k, inner, 0, n, inner, d, 1, 0);
  gemm(st, S.tln, d, 0, w.wv, inner, 0, S.v, inner, 0, n, inner, d, 1, 0);
  // query features: dd = (q*dn) @ proj^T  -> [h][n][m]
  gemm(st, S.q, inner, dh, proj, dh, 0, S.qp, mfeat, nm, n, mfeat, dh, h, GF_TB,
       nullptr, nullptr, 0, nullptr, 0, dn);
  k_row_sqsum<<<dim3(n, h), 256, 0, st>>>(S.q, inner, dh, S.diag, n, dh, 0.5f * dn * dn);
  k_row_max<<<dim3(n, h), 256, 0, st>>>(S.qp, mfeat, nm, S.stab, n, mfeat);
  k_feature_map<<<dim3((unsigned)((nm + 255) / 256), h), 256, 0, st>>>(
      S.qp, S.diag, S.stab, n, mfeat, ratio, 1, 1e-4f);
  // key features
  gemm(st, S.k, inner, dh, proj, dh, 0, S.kp, mfeat, nm, n, mfeat, dh, h, GF_TB,
       nullptr, nullptr, 0, nullptr, 0, dn);
  k_row_sqsum<<<dim3(n, h), 256, 0, st>>>(S.k, inner, dh, S.diag, n, dh, 0.5f * dn * dn);
  k_row_max<<<dim3(n, h), 256, 0, st>>>(S.kp, mfeat, nm, S.stab, n, mfeat);
  k_head_max<<<h, 256, 0, st>>>(S.stab, S.stabh, n);
  k_feature_map<<<dim3((unsigned)((nm + 255) / 256), h), 256, 0, st>>>(
      S.kp, S.diag, S.stabh, n, mfeat, ratio, 0, 1e-4f);
  // normalizer
  k_col_sum<<<dim3((mfeat + 255) / 256, h), 256, 0, st>>>(S.kp, S.ksum, n, mfeat);
  k_dinv<<<dim3(n, h), 256, 0, st>>>(S.qp, S.ksum, S.dinv, n, mfeat);
  // ctx[h][m][dh] = kp^T @ v
  gemm(st, S.kp, mfeat, nm, S.v, inner, dh, S.ctx, dh, (long)mfeat * dh,
       mfeat, dh, n, h, GF_TA);
  // o[n][h*dh] = (qp @ ctx) * dinv  (written directly into merged-head layout)
  gemm(st, S.qp, mfeat, nm, S.ctx, dh, (long)mfeat * dh, S.om, inner, dh,
       n, dh, mfeat, h, GF_RS, nullptr, nullptr, 0, S.dinv, n);
  // out projection + residual
  gemm(st, S.om, inner, 0, w.outw, d, 0, x, d, 0, n, d, inner, 1,
       GF_BIAS | GF_RESID, w.outb, x, 0);
  // FF block
  k_layernorm<<<n, 256, 0, st>>>(x, S.tln, w.ln2g, w.ln2b, n, d);
  gemm(st, S.tln, d, 0, w.ff1w, 4 * d, 0, S.ff, 4 * d, 0, n, 4 * d, d, 1,
       GF_BIAS | GF_GELU, w.ff1b);
  gemm(st, S.ff, 4 * d, 0, w.ff2w, d, 0, x, d, 0, n, d, 4 * d, 1,
       GF_BIAS | GF_RESID, w.ff2b, x, 0);
}

extern "C" void kernel_launch(void* const* d_in, const int* in_sizes, int n_in,
                              void* d_out, int out_size, void* d_ws, size_t ws_size,
                              hipStream_t stream) {
  (void)n_in; (void)out_size; (void)ws_size;
  const int N = 16906, L = 2048, BASE = 200, LARGE = 1280;
  const int E = in_sizes[2];
  auto P = [&](int i) { return (const float*)d_in[i]; };
  enum {
    I_X = 0, I_EI = 1, I_EW = 2, I_POS = 3, I_GO = 4, I_GOLW = 5, I_GOLB = 6, I_MASK = 7,
    I_FC1W = 8, I_FC1B = 9, I_FC2W = 10, I_FC2B = 11, I_TNG = 12, I_TNB = 13,
    I_B2LW = 14, I_B2LB = 15, I_LING = 16, I_LINB = 17,
    I_LL0 = 18,                 // 4 large layers x 13 arrays -> 18..69
    I_L2BW = 70, I_L2BB = 71, I_L2BNG = 72, I_L2BNB = 73,
    I_ML0 = 74,                 // mini layer -> 74..86
    I_DL0 = 87,                 // dec layer  -> 87..99
    I_DNG = 100, I_DNB = 101, I_EOW = 102, I_EOB = 103,
    I_PMINI = 104, I_PLARGE = 105, I_PDEC = 106
  };
  auto layerW = [&](int b) {
    return LayerW{P(b), P(b + 1), P(b + 2), P(b + 3), P(b + 4), P(b + 5), P(b + 6),
                  P(b + 7), P(b + 8), P(b + 9), P(b + 10), P(b + 11), P(b + 12)};
  };
  const int* ei = (const int*)d_in[I_EI];

  // ---- workspace bump allocator ----
  char* wsb = (char*)d_ws;
  size_t off = 0;
  auto alloc = [&](size_t bytes) {
    void* p = (void*)(wsb + off);
    off += (bytes + 255) & ~(size_t)255;
    return p;
  };
  float* x_emb  = (float*)alloc((size_t)N * BASE * 4);
  float* posgo  = (float*)alloc((size_t)N * BASE * 4);
  float* go_out = (float*)alloc((size_t)N * BASE * 4);
  float* go_agg = (float*)alloc((size_t)N * BASE * 4);
  float* deg    = (float*)alloc((size_t)N * 4);
  float* dis    = (float*)alloc((size_t)N * 4);
  unsigned* keys = (unsigned*)alloc((size_t)N * 4);
  unsigned* selTE = (unsigned*)alloc(64);
  int* top_idx  = (int*)alloc((size_t)L * 4);
  int* left_idx = (int*)alloc((size_t)(N - L) * 4);
  float* mrow   = (float*)alloc(256 * 4);
  float* xt     = (float*)alloc((size_t)L * LARGE * 4);
  float* xto    = (float*)alloc((size_t)L * BASE * 4);
  float* xl     = (float*)alloc((size_t)(N - L) * BASE * 4);
  float* merged = (float*)alloc((size_t)N * BASE * 4);
  const size_t tlnMax = (size_t)N * BASE;
  const size_t qkvMax = (size_t)N * 512;
  const size_t qpMax  = (size_t)8 * N * 266;   // >= 10*2048*621
  const size_t hnMax  = (size_t)10 * N;
  const size_t ffMax  = (size_t)N * 800;       // >= 2048*5120
  Scratch S;
  S.tln  = (float*)alloc(tlnMax * 4);
  S.q    = (float*)alloc(qkvMax * 4);
  S.k    = (float*)alloc(qkvMax * 4);
  S.v    = (float*)alloc(qkvMax * 4);
  S.qp   = (float*)alloc(qpMax * 4);
  S.kp   = (float*)alloc(qpMax * 4);
  S.diag = (float*)alloc(hnMax * 4);
  S.stab = (float*)alloc(hnMax * 4);
  S.stabh = (float*)alloc(64 * 4);
  S.ksum = (float*)alloc((size_t)10 * 621 * 4);
  S.dinv = (float*)alloc(hnMax * 4);
  S.ctx  = (float*)alloc((size_t)10 * 621 * 128 * 4);
  S.om   = (float*)alloc(qkvMax * 4);
  S.ff   = (float*)alloc(ffMax * 4);

  long nb = (long)N * BASE;
  // ---- embedding ----
  k_zero<<<(unsigned)((nb + 255) / 256), 256, 0, stream>>>(go_agg, nb);
  k_zero<<<(N + 255) / 256, 256, 0, stream>>>(deg, N);
  k_mask_renorm<<<1, 256, 0, stream>>>(P(I_MASK), mrow);
  k_token_embed<<<N, 256, 0, stream>>>(P(I_X), P(I_FC1W), P(I_FC1B), P(I_FC2W), P(I_FC2B),
                                       mrow, P(I_TNG), P(I_TNB), x_emb, N);
  k_deg<<<(E + 255) / 256, 256, 0, stream>>>(ei, P(I_EW), deg, E);
  k_dis<<<(N + 255) / 256, 256, 0, stream>>>(deg, dis, N);
  k_sgconv<<<E, 256, 0, stream>>>(ei, P(I_EW), dis, P(I_GO), go_agg, E);
  gemm(stream, go_agg, BASE, 0, P(I_GOLW), BASE, 0, go_out, BASE, 0,
       N, BASE, BASE, 1, GF_BIAS, P(I_GOLB));
  k_finish_embed<<<(unsigned)((nb + 255) / 256), 256, 0, stream>>>(P(I_POS), go_out, posgo, x_emb, nb);
  // ---- top-k split ----
  k_keys<<<(N + 255) / 256, 256, 0, stream>>>(P(I_X), keys, N);
  k_radix_select<<<1, 256, 0, stream>>>(keys, N, L, selTE);
  k_partition<<<1, 256, 0, stream>>>(keys, selTE, N, top_idx, left_idx);
  // ---- large path ----
  k_gather_rows<<<L, 256, 0, stream>>>(x_emb, top_idx, S.tln, L, BASE);
  gemm(stream, S.tln, BASE, 0, P(I_B2LW), LARGE, 0, xt, LARGE, 0,
       L, LARGE, BASE, 1, GF_BIAS, P(I_B2LB));
  k_layernorm<<<L, 256, 0, stream>>>(xt, xt, P(I_LING), P(I_LINB), L, LARGE);
  for (int l = 0; l < 4; ++l)
    performer_layer(stream, xt, L, LARGE, 10, 128, 621, P(I_PLARGE), layerW(I_LL0 + 13 * l), S);
  gemm(stream, xt, LARGE, 0, P(I_L2BW), BASE, 0, xto, BASE, 0,
       L, BASE, LARGE, 1, GF_BIAS, P(I_L2BB));
  k_layernorm<<<L, 256, 0, stream>>>(xto, xto, P(I_L2BNG), P(I_L2BNB), L, BASE);
  // ---- mini path ----
  int nL = N - L;
  k_gather_rows<<<nL, 256, 0, stream>>>(x_emb, left_idx, xl, nL, BASE);
  performer_layer(stream, xl, nL, BASE, 8, 64, 266, P(I_PMINI), layerW(I_ML0), S);
  // ---- merge + decode ----
  k_scatter_rows<<<L, 256, 0, stream>>>(xto, top_idx, merged, L, BASE);
  k_scatter_rows<<<nL, 256, 0, stream>>>(xl, left_idx, merged, nL, BASE);
  k_add<<<(unsigned)((nb + 255) / 256), 256, 0, stream>>>(merged, posgo, nb);
  performer_layer(stream, merged, N, BASE, 8, 64, 266, P(I_PDEC), layerW(I_DL0), S);
  k_layernorm<<<N, 256, 0, stream>>>(merged, merged, P(I_DNG), P(I_DNB), N, BASE);
  gemm(stream, merged, BASE, 0, P(I_EOW), 1, 0, (float*)d_out, 1, 0,
       N, 1, BASE, 1, GF_BIAS, P(I_EOB));
}
